// BinLinear_49203145343136
// MI455X (gfx1250) — 8-bit matrix operand order repaired and checked by reading, compiled, not run on hardware
//
#include <hip/hip_runtime.h>
#include <stdint.h>

typedef __attribute__((ext_vector_type(8))) int v8i;

#define M_DIM 16384
#define K_DIM 2048
#define N_DIM 2048
#define LN_EPS 1e-5f

#define BROW_STRIDE 80
#define BTILE_BYTES (128 * BROW_STRIDE)

__global__ void __launch_bounds__(256)
wbin_kernel(const float* __restrict__ weight, int8_t* __restrict__ Wb) {
    __shared__ float s_sum[256];
    const int row = blockIdx.x;
    const int t   = threadIdx.x;
    const float* wr = weight + (size_t)row * K_DIM;

    float4 v0 = ((const float4*)wr)[t * 2 + 0];
    float4 v1 = ((const float4*)wr)[t * 2 + 1];
    float s = v0.x + v0.y + v0.z + v0.w + v1.x + v1.y + v1.z + v1.w;
    s_sum[t] = s;
    __syncthreads();
    #pragma unroll
    for (int off = 128; off > 0; off >>= 1) {
        if (t < off) s_sum[t] += s_sum[t + off];
        __syncthreads();
    }
    const float mu = s_sum[0] * (1.0f / K_DIM);

    float xv[8] = {v0.x, v0.y, v0.z, v0.w, v1.x, v1.y, v1.z, v1.w};
    uint64_t pack = 0;
    #pragma unroll
    for (int i = 0; i < 8; ++i) {
        int8_t sgn = (xv[i] - mu) > 0.0f ? (int8_t)1 : (int8_t)-1;
        pack |= ((uint64_t)(uint8_t)sgn) << (8 * i);
    }
    ((uint64_t*)(Wb + (size_t)row * K_DIM))[t] = pack;
}

__global__ void __launch_bounds__(256)
ln_sign_kernel(const float* __restrict__ x,
               const float* __restrict__ gamma,
               const float* __restrict__ beta,
               int8_t* __restrict__ Ab) {
    __shared__ float s_sum[256];
    __shared__ float s_sq[256];
    const int row = blockIdx.x;
    const int t   = threadIdx.x;
    const float* xr = x + (size_t)row * K_DIM;

    float4 v0 = ((const float4*)xr)[t * 2 + 0];
    float4 v1 = ((const float4*)xr)[t * 2 + 1];
    float s = v0.x + v0.y + v0.z + v0.w + v1.x + v1.y + v1.z + v1.w;
    float q = v0.x * v0.x + v0.y * v0.y + v0.z * v0.z + v0.w * v0.w
            + v1.x * v1.x + v1.y * v1.y + v1.z * v1.z + v1.w * v1.w;
    s_sum[t] = s;
    s_sq[t]  = q;
    __syncthreads();
    #pragma unroll
    for (int off = 128; off > 0; off >>= 1) {
        if (t < off) { s_sum[t] += s_sum[t + off]; s_sq[t] += s_sq[t + off]; }
        __syncthreads();
    }
    const float mu  = s_sum[0] * (1.0f / K_DIM);
    const float var = s_sq[0] * (1.0f / K_DIM) - mu * mu;
    const float rs  = rsqrtf(var + LN_EPS);

    float xv[8] = {v0.x, v0.y, v0.z, v0.w, v1.x, v1.y, v1.z, v1.w};
    const int base = t * 8;
    uint64_t pack = 0;
    #pragma unroll
    for (int i = 0; i < 8; ++i) {
        float xn = (xv[i] - mu) * rs * gamma[base + i] + beta[base + i];
        int8_t sgn = xn > 0.0f ? (int8_t)1 : (int8_t)-1;
        pack |= ((uint64_t)(uint8_t)sgn) << (8 * i);
    }
    ((uint64_t*)(Ab + (size_t)row * K_DIM))[t] = pack;
}

__global__ void __launch_bounds__(256)
bingemm_kernel(const int8_t* __restrict__ Ab,
               const int8_t* __restrict__ Wb,
               const float* __restrict__ bias,
               const float* __restrict__ alpha,
               float* __restrict__ out) {
    __shared__ alignas(16) int8_t smemB[2][BTILE_BYTES];

    const int t    = threadIdx.x;
    const int lane = t & 31;
    const int wave = t >> 5;
    const int half = lane >> 4;
    const int l16  = lane & 15;

    const int m0 = blockIdx.y * 128 + wave * 16;
    const int n0 = blockIdx.x * 128;

    const int8_t* aBase = Ab + (size_t)(m0 + l16) * K_DIM + half * 32;

    auto issue_btile = [&](int buf, int k) {
        #pragma unroll
        for (int i = 0; i < 2; ++i) {
            const int ch  = t * 2 + i;
            const int c   = ch >> 2;
            const int off = (ch & 3) * 16;
            const uint64_t gaddr = (uint64_t)(uintptr_t)
                (Wb + (size_t)(n0 + c) * K_DIM + k + off);
            const uint32_t laddr = (uint32_t)(uintptr_t)
                (&smemB[buf][c * BROW_STRIDE + off]);
            asm volatile("global_load_async_to_lds_b128 %0, %1, off"
                         :: "v"(laddr), "v"(gaddr)
                         : "memory");
        }
    };

    v8i acc[8];
    #pragma unroll
    for (int g = 0; g < 8; ++g) acc[g] = (v8i){0, 0, 0, 0, 0, 0, 0, 0};

    issue_btile(0, 0);
    asm volatile("s_wait_asynccnt 0x0" ::: "memory");
    __syncthreads();

    union BF { int4 q[2]; v8i v; };
    const int bLane = l16 * BROW_STRIDE + half * 32;

    int cur = 0;
    for (int k = 0; k < K_DIM; k += 64) {
        if (k + 64 < K_DIM) issue_btile(cur ^ 1, k + 64);

        union { uint64_t u[4]; v8i v; } af;
        #pragma unroll
        for (int i = 0; i < 4; ++i)
            af.u[i] = *(const uint64_t*)(aBase + k + 8 * i);

        const int8_t* bTile = &smemB[cur][0];

        BF bcur, bnext;
        {
            const int4* bp = (const int4*)(bTile + bLane);
            bcur.q[0] = bp[0];
            bcur.q[1] = bp[1];
        }
        #pragma unroll
        for (int g = 0; g < 8; ++g) {
            if (g < 7) {
                const int4* bp =
                    (const int4*)(bTile + (g + 1) * 16 * BROW_STRIDE + bLane);
                bnext.q[0] = bp[0];
                bnext.q[1] = bp[1];
            }
            acc[g] = __builtin_amdgcn_wmma_i32_16x16x64_iu8(
                true, af.v, true, bcur.v, acc[g], false, false);
            bcur = bnext;
        }

        asm volatile("s_wait_asynccnt 0x0" ::: "memory");
        __syncthreads();
        cur ^= 1;
    }

    #pragma unroll
    for (int g = 0; g < 8; ++g) {
        const int n = n0 + g * 16 + l16;
        const float b  = bias[n];
        const float al = alpha[n];
        #pragma unroll
        for (int j = 0; j < 8; ++j) {
            const int m = m0 + j + half * 8;
            out[(size_t)m * N_DIM + n] = ((float)acc[g][j] + b) * al;
        }
    }
}

extern "C" void kernel_launch(void* const* d_in, const int* in_sizes, int n_in,
                              void* d_out, int out_size, void* d_ws, size_t ws_size,
                              hipStream_t stream) {
    const float* x      = (const float*)d_in[0];
    const float* gamma  = (const float*)d_in[1];
    const float* beta   = (const float*)d_in[2];
    const float* weight = (const float*)d_in[3];
    const float* bias   = (const float*)d_in[4];
    const float* alpha  = (const float*)d_in[5];
    float* out = (float*)d_out;

    int8_t* Wb = (int8_t*)d_ws;
    int8_t* Ab = (int8_t*)d_ws + (size_t)N_DIM * K_DIM;

    wbin_kernel<<<N_DIM, 256, 0, stream>>>(weight, Wb);
    ln_sign_kernel<<<M_DIM, 256, 0, stream>>>(x, gamma, beta, Ab);

    dim3 grid(N_DIM / 128, M_DIM / 128);
    bingemm_kernel<<<grid, 256, 0, stream>>>(Ab, Wb, bias, alpha, out);
}
